// MambaBlock_34780645163484
// MI455X (gfx1250) — matrix operand order repaired and checked by reading, not compiled, not run on hardware
//
#include <hip/hip_runtime.h>
#include <hip/hip_bf16.h>
#include <math.h>

// Problem constants (from the reference)
#define DM   256      // d_model
#define DI   512      // d_inner
#define NS   64       // n_state
#define DTR  16       // dt_rank
#define KC   4        // conv kernel size
#define BB   2        // batch
#define LL   1024     // seq len
#define ML   (BB*LL)  // flattened rows = 2048
#define XDBL_N (DTR + 2*NS)   // 144
#define DTR_PAD 32            // dt_rank zero-padded to one K-step

typedef __bf16 bf16_t;
typedef __attribute__((ext_vector_type(16))) __bf16 v16bf;
typedef __attribute__((ext_vector_type(8)))  __bf16 v8bf;
typedef __attribute__((ext_vector_type(8)))  float  v8f;

// ---------------------------------------------------------------------------
// bf16 WMMA GEMM, compile-time K: C[M,N] = A[M,K] * Bt[N,K]^T (+epilogue)
// A is (M,K) bf16 row-major; Bt is (N,K) bf16 row-major (i.e. B transposed),
// so both fragments are contiguous vector loads:
//   A frag : lane(m = r) k = k0 + 16*half + {0..7, 8..15}  -> 2x 16B loads
//   B frag : lane(n = r) k = k0 + 16*half + {0..15}        -> 1x 32B load
// One wave per 16x16 tile; no divergence anywhere (EXEC all-1s at WMMA).
// EP: 0 = none; 1 = softplus(v + bias[n])
// ---------------------------------------------------------------------------
template<int K, int EP>
__global__ __launch_bounds__(128) void gemm_wmma_bf16(
    const bf16_t* __restrict__ A,
    const bf16_t* __restrict__ Bt,
    float* __restrict__ C,
    int M, int N,
    const float* __restrict__ bias)
{
  const int wave = blockIdx.x * (blockDim.x >> 5) + (threadIdx.x >> 5);
  const int lane = threadIdx.x & 31;
  const int tilesN   = N >> 4;
  const int numTiles = (M >> 4) * tilesN;
  if (wave >= numTiles) return;            // wave-uniform guard

  const int tm   = (wave / tilesN) << 4;
  const int tn   = (wave % tilesN) << 4;
  const int half = lane >> 4;
  const int r    = lane & 15;

  const bf16_t* Arow = A  + (size_t)(tm + r) * K + 16 * half;
  const bf16_t* Brow = Bt + (size_t)(tn + r) * K + 16 * half;

  v8f acc = {};
#pragma unroll 4
  for (int k0 = 0; k0 < K; k0 += 32) {
    v8bf alo = *(const v8bf*)(Arow + k0);        // k0+16h+0..7
    v8bf ahi = *(const v8bf*)(Arow + k0 + 8);    // k0+16h+8..15
    v16bf a = __builtin_shufflevector(alo, ahi,
        0,1,2,3,4,5,6,7,8,9,10,11,12,13,14,15);
    v16bf b = *(const v16bf*)(Brow + k0);        // k0+16h+0..15
    acc = __builtin_amdgcn_wmma_f32_16x16x32_bf16(
        false, a, false, b, (short)0, acc, false, false);
  }

#pragma unroll
  for (int j = 0; j < 8; ++j) {
    int m = tm + j + 8 * half;                   // C/D: VGPR j -> row j+8*half
    int n = tn + r;
    float v = acc[j];
    if (EP == 1) {
      v += bias[n];
      v = (v > 20.0f) ? v : log1pf(__expf(v));   // softplus
    }
    C[(size_t)m * N + n] = v;
  }
}

// ---------------------------------------------------------------------------
// Weight prep: out[n*Kp + k] = bf16(in[k*N + n]), zero-padded for k >= K.
// ---------------------------------------------------------------------------
__global__ __launch_bounds__(256) void transpose_to_bf16(
    const float* __restrict__ in, bf16_t* __restrict__ out,
    int K, int N, int Kp)
{
  int idx = blockIdx.x * blockDim.x + threadIdx.x;
  if (idx >= N * Kp) return;
  int k = idx % Kp, n = idx / Kp;
  out[idx] = (k < K) ? (bf16_t)in[(size_t)k * N + n] : (bf16_t)0.0f;
}

__global__ __launch_bounds__(256) void f32_to_bf16_kernel(
    const float* __restrict__ in, bf16_t* __restrict__ out, int n)
{
  int idx = blockIdx.x * blockDim.x + threadIdx.x;
  if (idx < n) out[idx] = (bf16_t)in[idx];
}

// Extract dt-rank slice of x_dbl as bf16, padded (ML x DTR_PAD).
__global__ __launch_bounds__(256) void extract_dl_kernel(
    const float* __restrict__ xdbl, bf16_t* __restrict__ dl)
{
  int idx = blockIdx.x * blockDim.x + threadIdx.x;
  if (idx >= ML * DTR_PAD) return;
  int k = idx % DTR_PAD, m = idx / DTR_PAD;
  dl[idx] = (k < DTR) ? (bf16_t)xdbl[(size_t)m * XDBL_N + k] : (bf16_t)0.0f;
}

// ---------------------------------------------------------------------------
// Causal depthwise conv1d (K=4, left pad 3) + SiLU. Writes f32 (for scan)
// and bf16 (A operand of the x_dbl GEMM).
// ---------------------------------------------------------------------------
__global__ __launch_bounds__(256) void conv_swish_kernel(
    const float* __restrict__ xr,     // (ML, 2*DI)
    const float* __restrict__ ck,     // (KC, DI)
    const float* __restrict__ cb,     // (DI)
    float* __restrict__ xc,           // (ML, DI)
    bf16_t* __restrict__ xc_bf)       // (ML, DI)
{
  int idx = blockIdx.x * blockDim.x + threadIdx.x;
  if (idx >= ML * DI) return;
  int d = idx % DI;
  int l = (idx / DI) % LL;
  int b = idx / (DI * LL);
  float acc = cb[d];
#pragma unroll
  for (int j = 0; j < KC; ++j) {
    int ls = l - (KC - 1) + j;
    if (ls >= 0)
      acc += xr[((size_t)(b * LL + ls)) * (2 * DI) + d] * ck[j * DI + d];
  }
  float s = acc / (1.0f + __expf(-acc));   // swish
  xc[idx] = s;
  xc_bf[idx] = (bf16_t)s;
}

// ---------------------------------------------------------------------------
// Selective scan fused with skip (u*D) and swish(res) gating; emits bf16 y
// (A operand of the output GEMM). The reference's flip/cumsum/exp/flip is
// algebraically: h_t = exp(dt*A) h_{t-1} + dt*u*B_t ; y_t = <h_t, C_t>.
// One wave32 per (b, d); each lane owns states n=lane and n=lane+32.
// ---------------------------------------------------------------------------
__global__ __launch_bounds__(128) void scan_kernel(
    const float* __restrict__ xc,     // (ML, DI)  u
    const float* __restrict__ delta,  // (ML, DI)
    const float* __restrict__ xdbl,   // (ML, 144) [dl | B | C]
    const float* __restrict__ A_log,  // (DI, NS)
    const float* __restrict__ Dp,     // (DI)
    const float* __restrict__ xr,     // (ML, 2*DI) res = cols [DI, 2*DI)
    bf16_t* __restrict__ y_bf)        // (ML, DI)
{
  int wave = (blockIdx.x * blockDim.x + threadIdx.x) >> 5;
  int lane = threadIdx.x & 31;
  if (wave >= BB * DI) return;
  int b = wave / DI;
  int d = wave % DI;

  const int n0 = lane, n1 = lane + 32;
  const float A0 = -__expf(A_log[d * NS + n0]);
  const float A1 = -__expf(A_log[d * NS + n1]);
  const float Dv = Dp[d];

  float h0 = 0.0f, h1 = 0.0f;
  for (int l = 0; l < LL; ++l) {
    const int bl = b * LL + l;
    const float dt = delta[(size_t)bl * DI + d];    // uniform across wave
    const float u  = xc[(size_t)bl * DI + d];
    const float* row = xdbl + (size_t)bl * XDBL_N;
    const float Bm0 = row[DTR + n0],      Bm1 = row[DTR + n1];
    const float Cm0 = row[DTR + NS + n0], Cm1 = row[DTR + NS + n1];

    h0 = __expf(dt * A0) * h0 + dt * u * Bm0;
    h1 = __expf(dt * A1) * h1 + dt * u * Bm1;

    float part = h0 * Cm0 + h1 * Cm1;
#pragma unroll
    for (int off = 16; off > 0; off >>= 1)
      part += __shfl_xor(part, off, 32);

    if (lane == 0) {
      float res = xr[(size_t)bl * (2 * DI) + DI + d];
      float g = res / (1.0f + __expf(-res));        // swish(res)
      y_bf[(size_t)bl * DI + d] = (bf16_t)((part + u * Dv) * g);
    }
  }
}

// ---------------------------------------------------------------------------
// Launch pipeline.
// ---------------------------------------------------------------------------
extern "C" void kernel_launch(void* const* d_in, const int* in_sizes, int n_in,
                              void* d_out, int out_size, void* d_ws, size_t ws_size,
                              hipStream_t stream)
{
  (void)in_sizes; (void)n_in; (void)out_size; (void)ws_size;
  const float* x     = (const float*)d_in[0];
  const float* W_in  = (const float*)d_in[1];
  const float* ck    = (const float*)d_in[2];
  const float* cb    = (const float*)d_in[3];
  const float* W_x   = (const float*)d_in[4];
  const float* W_dt  = (const float*)d_in[5];
  const float* b_dt  = (const float*)d_in[6];
  const float* A_log = (const float*)d_in[7];
  const float* Dp    = (const float*)d_in[8];
  const float* W_out = (const float*)d_in[9];
  float* out = (float*)d_out;

  // Workspace carve (all chunk sizes are multiples of 256 bytes).
  char* p = (char*)d_ws;
  float*  xr     = (float*)p;  p += (size_t)ML * 2 * DI * 4;   // 8 MB
  float*  xc     = (float*)p;  p += (size_t)ML * DI * 4;       // 4 MB
  float*  xdbl   = (float*)p;  p += (size_t)ML * XDBL_N * 4;   // 1.2 MB
  float*  delta  = (float*)p;  p += (size_t)ML * DI * 4;       // 4 MB
  bf16_t* x_bf   = (bf16_t*)p; p += (size_t)ML * DM * 2;       // 1 MB
  bf16_t* xc_bf  = (bf16_t*)p; p += (size_t)ML * DI * 2;       // 2 MB
  bf16_t* dl_bf  = (bf16_t*)p; p += (size_t)ML * DTR_PAD * 2;  // 128 KB
  bf16_t* y_bf   = (bf16_t*)p; p += (size_t)ML * DI * 2;       // 2 MB
  bf16_t* Wt_in  = (bf16_t*)p; p += (size_t)(2*DI) * DM * 2;   // (1024,256)
  bf16_t* Wt_x   = (bf16_t*)p; p += (size_t)XDBL_N * DI * 2;   // (144,512)
  bf16_t* Wt_dt  = (bf16_t*)p; p += (size_t)DI * DTR_PAD * 2;  // (512,32)
  bf16_t* Wt_out = (bf16_t*)p; p += (size_t)DM * DI * 2;       // (256,512)

  // --- operand prep (bf16, transposed weights) ---
  f32_to_bf16_kernel<<<(ML*DM + 255)/256, 256, 0, stream>>>(x, x_bf, ML*DM);
  transpose_to_bf16<<<((2*DI)*DM + 255)/256, 256, 0, stream>>>(W_in,  Wt_in,  DM,  2*DI,  DM);
  transpose_to_bf16<<<(XDBL_N*DI + 255)/256, 256, 0, stream>>>(W_x,   Wt_x,   DI,  XDBL_N, DI);
  transpose_to_bf16<<<(DI*DTR_PAD + 255)/256, 256, 0, stream>>>(W_dt,  Wt_dt,  DTR, DI,    DTR_PAD);
  transpose_to_bf16<<<(DM*DI + 255)/256, 256, 0, stream>>>(W_out, Wt_out, DI,  DM,    DI);

  // 1. x_and_res = x @ W_in                    (2048 x 256 x 1024)
  {
    int tiles = (ML/16) * ((2*DI)/16);
    gemm_wmma_bf16<DM, 0><<<(tiles + 3)/4, 128, 0, stream>>>(
        x_bf, Wt_in, xr, ML, 2*DI, nullptr);
  }
  // 2. xc = swish(depthwise_conv(x_int) + conv_b)  (f32 + bf16)
  conv_swish_kernel<<<(ML*DI + 255)/256, 256, 0, stream>>>(xr, ck, cb, xc, xc_bf);
  // 3. x_dbl = xc @ W_x                        (2048 x 512 x 144)
  {
    int tiles = (ML/16) * (XDBL_N/16);
    gemm_wmma_bf16<DI, 0><<<(tiles + 3)/4, 128, 0, stream>>>(
        xc_bf, Wt_x, xdbl, ML, XDBL_N, nullptr);
  }
  // 3b. dl slice -> bf16, zero-padded to K=32
  extract_dl_kernel<<<(ML*DTR_PAD + 255)/256, 256, 0, stream>>>(xdbl, dl_bf);
  // 4. delta = softplus(dl @ W_dt + b_dt)      (2048 x 32 x 512)
  {
    int tiles = (ML/16) * (DI/16);
    gemm_wmma_bf16<DTR_PAD, 1><<<(tiles + 3)/4, 128, 0, stream>>>(
        dl_bf, Wt_dt, delta, ML, DI, b_dt);
  }
  // 5. selective scan + u*D + swish(res) gating -> bf16 y
  scan_kernel<<<(BB*DI*32 + 127)/128, 128, 0, stream>>>(
      xc, delta, xdbl, A_log, Dp, xr, y_bf);
  // 6. out = y @ W_out                          (2048 x 512 x 256)
  {
    int tiles = (ML/16) * (DM/16);
    gemm_wmma_bf16<DI, 0><<<(tiles + 3)/4, 128, 0, stream>>>(
        y_bf, Wt_out, out, ML, DM, nullptr);
  }
}
